// MyAttention_45372034515024
// MI455X (gfx1250) — compile-verified, matrix operand order checked by reading, not run on hardware
//
#include <hip/hip_runtime.h>
#include <hip/hip_bf16.h>

// ---------------------------------------------------------------------------
// Squeezed-KV attention for MI455X (gfx1250, wave32, WMMA).
//   B=16, L=S=4096, H=16, D=64, M=256
//   k1: K_sq/V_sq = W @ K/V (+bias) -> bf16 ws (K_sq row-major, V_sq transposed)
//   k2: S = Q K_sq^T, softmax(m), O = P V_sq   (flash-style, M=256 in regs)
// All WMMA fragments are 2x ds_load_b128; kernel-2 tile staging uses
// GLOBAL_LOAD_ASYNC_TO_LDS_B128 (ASYNCcnt) instead of VGPR round-trips.
// ---------------------------------------------------------------------------

typedef __attribute__((ext_vector_type(16))) __bf16 v16bf;
typedef __attribute__((ext_vector_type(8)))  float  v8f;

#define WMMA_BF16(a, b, c) \
  __builtin_amdgcn_wmma_f32_16x16x32_bf16(false, (a), false, (b), (short)0, (c), false, false)

__device__ __forceinline__ v8f vzero8() {
  v8f z = {0.f, 0.f, 0.f, 0.f, 0.f, 0.f, 0.f, 0.f};
  return z;
}

// Fragment load from k-innermost LDS tile. `p` already includes the lane's
// row offset and the (lane&16 ? 8 : 0) k-shift. Elements 0..7 = p[0..7],
// elements 8..15 = p[16..23]  (ISA 7.12.2 16-bit A/B layout). 16B aligned.
__device__ __forceinline__ v16bf ld_frag(const __bf16* p) {
  struct alignas(32) P2 { uint4 x, y; } t;
  t.x = *(const uint4*)(p);
  t.y = *(const uint4*)(p + 16);
  return __builtin_bit_cast(v16bf, t);
}

// pack two f32 -> packed bf16x2 (for b32 LDS staging stores)
__device__ __forceinline__ unsigned int pack2bf(float lo, float hi) {
  unsigned short a = __builtin_bit_cast(unsigned short, (__bf16)lo);
  unsigned short b = __builtin_bit_cast(unsigned short, (__bf16)hi);
  return (unsigned int)a | ((unsigned int)b << 16);
}

// build bf16 A-fragment from 4 float4 (elements 0..7 = a,b ; 8..15 = c,d)
__device__ __forceinline__ v16bf cvt_frag(float4 a, float4 b, float4 c, float4 d) {
  v16bf r;
  r[0] = (__bf16)a.x;  r[1] = (__bf16)a.y;  r[2] = (__bf16)a.z;  r[3] = (__bf16)a.w;
  r[4] = (__bf16)b.x;  r[5] = (__bf16)b.y;  r[6] = (__bf16)b.z;  r[7] = (__bf16)b.w;
  r[8] = (__bf16)c.x;  r[9] = (__bf16)c.y;  r[10] = (__bf16)c.z; r[11] = (__bf16)c.w;
  r[12] = (__bf16)d.x; r[13] = (__bf16)d.y; r[14] = (__bf16)d.z; r[15] = (__bf16)d.w;
  return r;
}

// async global -> LDS copy, 16 bytes per lane (GVS mode: saddr + 32-bit voff)
__device__ __forceinline__ void async_copy_b128(unsigned int lds_addr,
                                                unsigned int voff,
                                                unsigned long long gbase) {
  asm volatile("global_load_async_to_lds_b128 %0, %1, %2 offset:0"
               :: "v"(lds_addr), "v"(voff), "s"(gbase) : "memory");
}

// ---------------------------------------------------------------------------
// Kernel 1: per (b,h): Out[m,d] = sum_s W[m,s] * X[b,s,h,d] + bias[m]
//   grid = 256 blocks, 256 threads (8 waves); wave owns 2 m-tiles.
//   K_sq stored [bh][m][d]; V_sq stored transposed [bh][d][m] for kernel 2.
// ---------------------------------------------------------------------------
__global__ __launch_bounds__(256, 1)
void squeeze_kv_kernel(const float* __restrict__ keys,
                       const float* __restrict__ values,
                       const float* __restrict__ sqw,
                       const float* __restrict__ sqb,
                       __bf16* __restrict__ ksq,
                       __bf16* __restrict__ vsqT) {
  const int bh   = blockIdx.x;
  const int b    = bh >> 4;
  const int h    = bh & 15;
  const int tid  = threadIdx.x;
  const int w    = tid >> 5;
  const int lane = tid & 31;
  const int ln16 = lane & 15;
  const int hi8  = (lane & 16) ? 8 : 0;

  __shared__ __bf16 sW[256 * 32];   // [m][s]  k-innermost for A-frags
  __shared__ __bf16 sKt[64 * 32];   // [d][s]  k-innermost for B-frags
  __shared__ __bf16 sVt[64 * 32];   // [d][s]

  v8f accK[2][4], accV[2][4];
#pragma unroll
  for (int mt = 0; mt < 2; ++mt)
#pragma unroll
    for (int n = 0; n < 4; ++n) { accK[mt][n] = vzero8(); accV[mt][n] = vzero8(); }

  for (int s0 = 0; s0 < 4096; s0 += 32) {
    // W tile: float2 global reads -> packed bf16x2 b32 LDS stores
    for (int i = tid; i < 4096; i += 256) {          // 4096 bf16-pairs
      int m = i >> 4, sp = (i & 15) << 1;
      const float2 g = *(const float2*)(sqw + m * 4096 + s0 + sp);
      ((unsigned int*)sW)[i] = pack2bf(g.x, g.y);    // sW[m*32 + sp .. +1]
    }
    // K/V tiles: coalesced global b32 reads, transposed b16 LDS writes
    for (int i = tid; i < 2048; i += 256) {
      int s = i >> 6, d = i & 63;
      long g = (((long)(b * 4096 + s0 + s) * 16 + h) * 64 + d);
      sKt[d * 32 + s] = (__bf16)keys[g];
      sVt[d * 32 + s] = (__bf16)values[g];
    }
    __syncthreads();

    const v16bf a0 = ld_frag(sW + (w * 32 + ln16) * 32 + hi8);
    const v16bf a1 = ld_frag(sW + (w * 32 + 16 + ln16) * 32 + hi8);
#pragma unroll
    for (int n = 0; n < 4; ++n) {
      const v16bf bk = ld_frag(sKt + (n * 16 + ln16) * 32 + hi8);
      const v16bf bv = ld_frag(sVt + (n * 16 + ln16) * 32 + hi8);
      accK[0][n] = WMMA_BF16(a0, bk, accK[0][n]);
      accK[1][n] = WMMA_BF16(a1, bk, accK[1][n]);
      accV[0][n] = WMMA_BF16(a0, bv, accV[0][n]);
      accV[1][n] = WMMA_BF16(a1, bv, accV[1][n]);
    }
    __syncthreads();
  }

  // epilogue: +bias, store K_sq [m][d] and V_sq transposed [d][m]
  const long obase = (long)bh * 256 * 64;
#pragma unroll
  for (int mt = 0; mt < 2; ++mt) {
#pragma unroll
    for (int v = 0; v < 8; ++v) {
      int m = w * 32 + mt * 16 + v + hi8;
      float bias = sqb[m];
#pragma unroll
      for (int n = 0; n < 4; ++n) {
        int d = n * 16 + ln16;
        ksq[obase + (long)m * 64 + d]  = (__bf16)(accK[mt][n][v] + bias);
        vsqT[obase + (long)d * 256 + m] = (__bf16)(accV[mt][n][v] + bias);
      }
    }
  }
}

// ---------------------------------------------------------------------------
// Kernel 2: attention. grid = B*H*(L/128) = 8192 blocks, 256 threads.
//   LDS (dynamic, 128 KB): sK [m][d] 32K + sVt [d][m] 32K + per-wave P 8x8K.
//   K_sq/V_sqT tiles staged with GLOBAL_LOAD_ASYNC_TO_LDS_B128 (ASYNCcnt).
// ---------------------------------------------------------------------------
__global__ __launch_bounds__(256, 1)
void attn_kernel(const float* __restrict__ q,
                 const __bf16* __restrict__ ksq,
                 const __bf16* __restrict__ vsqT,
                 float* __restrict__ out) {
  extern __shared__ __bf16 smem[];
  __bf16* sK  = smem;                        // [m][d] 256x64 (k=d innermost)
  __bf16* sVt = smem + 16384;                // [d][m] 64x256 (k=m innermost)
  const int tid  = threadIdx.x;
  const int w    = tid >> 5;
  const int lane = tid & 31;
  const int ln16 = lane & 15;
  const int hi8  = (lane & 16) ? 8 : 0;
  __bf16* sP = smem + 32768 + w * (16 * 256); // per-wave 16x256 P tile

  const int lb = blockIdx.x & 31;
  const int bh = blockIdx.x >> 5;
  const int b  = bh >> 4;
  const int h  = bh & 15;

  // stage K_sq / V_sqT slices (32 KB each): async global->LDS, no VGPR bounce
  {
    const unsigned long long gka = (unsigned long long)(ksq + (long)bh * 16384);
    const unsigned long long gva = (unsigned long long)(vsqT + (long)bh * 16384);
#pragma unroll
    for (int i = tid; i < 2048; i += 256) {
      unsigned int voff = (unsigned int)i * 16u;   // byte offset into slice
      unsigned int dk = (unsigned int)(size_t)(sK + i * 8);
      unsigned int dv = (unsigned int)(size_t)(sVt + i * 8);
      async_copy_b128(dk, voff, gka);
      async_copy_b128(dv, voff, gva);
    }
    asm volatile("s_wait_asynccnt 0" ::: "memory");
  }
  __syncthreads();

  const int r0 = lb * 128 + w * 16;

  // Q A-fragments via float4 loads (k = d: 0..31 and 32..63)
  v16bf aq0, aq1;
  {
    const float* qp = q + (((long)(b * 4096 + r0 + ln16)) * 16 + h) * 64;
    float4 f0 = *(const float4*)(qp + hi8);
    float4 f1 = *(const float4*)(qp + hi8 + 4);
    float4 f2 = *(const float4*)(qp + hi8 + 16);
    float4 f3 = *(const float4*)(qp + hi8 + 20);
    aq0 = cvt_frag(f0, f1, f2, f3);
    f0 = *(const float4*)(qp + 32 + hi8);
    f1 = *(const float4*)(qp + 32 + hi8 + 4);
    f2 = *(const float4*)(qp + 32 + hi8 + 16);
    f3 = *(const float4*)(qp + 32 + hi8 + 20);
    aq1 = cvt_frag(f0, f1, f2, f3);
  }

  // S = Q K_sq^T : 16 n-tiles over m, 2 wmma each (all frag loads = b128)
  v8f acc[16];
#pragma unroll
  for (int t = 0; t < 16; ++t) acc[t] = vzero8();
#pragma unroll
  for (int t = 0; t < 16; ++t) {
    const v16bf b0 = ld_frag(sK + (t * 16 + ln16) * 64 + hi8);
    const v16bf b1 = ld_frag(sK + (t * 16 + ln16) * 64 + 32 + hi8);
    acc[t] = WMMA_BF16(aq0, b0, acc[t]);
    acc[t] = WMMA_BF16(aq1, b1, acc[t]);
  }

  // softmax over m (C-layout: row = v + hi8, col = t*16 + ln16)
  const float scale = 0.125f;  // 1/sqrt(64)
  float mx[8], sm[8], inv[8];
#pragma unroll
  for (int v = 0; v < 8; ++v) mx[v] = -3.0e38f;
#pragma unroll
  for (int t = 0; t < 16; ++t)
#pragma unroll
    for (int v = 0; v < 8; ++v) mx[v] = fmaxf(mx[v], acc[t][v]);
#pragma unroll
  for (int d = 1; d < 16; d <<= 1)    // half-wave reduce keeps row halves split
#pragma unroll
    for (int v = 0; v < 8; ++v) mx[v] = fmaxf(mx[v], __shfl_xor(mx[v], d, 32));
#pragma unroll
  for (int v = 0; v < 8; ++v) sm[v] = 0.f;
#pragma unroll
  for (int t = 0; t < 16; ++t)
#pragma unroll
    for (int v = 0; v < 8; ++v) {
      float p = __expf((acc[t][v] - mx[v]) * scale);
      acc[t][v] = p;
      sm[v] += p;
    }
#pragma unroll
  for (int d = 1; d < 16; d <<= 1)
#pragma unroll
    for (int v = 0; v < 8; ++v) sm[v] += __shfl_xor(sm[v], d, 32);
#pragma unroll
  for (int v = 0; v < 8; ++v) inv[v] = 1.0f / sm[v];

  // write normalized P tile (16x256) once: C-layout -> row-major LDS
#pragma unroll
  for (int t = 0; t < 16; ++t)
#pragma unroll
    for (int v = 0; v < 8; ++v)
      sP[(v + hi8) * 256 + t * 16 + ln16] = (__bf16)(acc[t][v] * inv[v]);
  asm volatile("s_wait_dscnt 0" ::: "memory");   // same-wave LDS RAW

  // O = P V_sq : K=256 in 8 chunks; A-frags and B-frags are b128 loads
  v8f o[4];
#pragma unroll
  for (int n = 0; n < 4; ++n) o[n] = vzero8();
#pragma unroll
  for (int ks = 0; ks < 8; ++ks) {
    const v16bf ap = ld_frag(sP + ln16 * 256 + ks * 32 + hi8);
#pragma unroll
    for (int n = 0; n < 4; ++n) {
      const v16bf bv = ld_frag(sVt + (n * 16 + ln16) * 256 + ks * 32 + hi8);
      o[n] = WMMA_BF16(ap, bv, o[n]);
    }
  }

  // store context f32 [b][l][h][d] (coalesced across ln16)
#pragma unroll
  for (int n = 0; n < 4; ++n)
#pragma unroll
    for (int v = 0; v < 8; ++v) {
      int row = r0 + v + hi8;
      out[(((long)(b * 4096 + row)) * 16 + h) * 64 + n * 16 + ln16] = o[n][v];
    }
}

// ---------------------------------------------------------------------------
extern "C" void kernel_launch(void* const* d_in, const int* in_sizes, int n_in,
                              void* d_out, int out_size, void* d_ws, size_t ws_size,
                              hipStream_t stream) {
  const float* queries = (const float*)d_in[0];  // (16,4096,16,64)
  const float* keys    = (const float*)d_in[1];  // (16,4096,16,64)
  const float* values  = (const float*)d_in[2];  // (16,4096,16,64)
  const float* sqw     = (const float*)d_in[3];  // (256,4096)
  const float* sqb     = (const float*)d_in[4];  // (256,)
  float* out = (float*)d_out;                    // (16,4096,16,64) f32

  __bf16* ksq  = (__bf16*)d_ws;                          // 8 MB [bh][m][d]
  __bf16* vsqT = ksq + (size_t)16 * 16 * 256 * 64;       // 8 MB [bh][d][m]

  squeeze_kv_kernel<<<dim3(256), dim3(256), 0, stream>>>(keys, values, sqw, sqb, ksq, vsqT);
  // 128 KB dynamic LDS: 32K sK + 32K sVt + 8x8K per-wave P (gfx1250: 320 KB/WGP)
  attn_kernel<<<dim3(8192), dim3(256), 128 * 1024, stream>>>(queries, ksq, vsqT, out);
}
